// UserEncode_44839458570801
// MI455X (gfx1250) — matrix operand order repaired and checked by reading, not compiled, not run on hardware
//
#include <hip/hip_runtime.h>

// ---------------------------------------------------------------------------
// GraphRec user-encoder for MI455X (gfx1250, wave32, WMMA).
// One node per workgroup, 128 threads = 4 waves; each wave computes a 16-row
// stripe of the padded (64 x *) activation matrices with v_wmma_f32_16x16x32_f16.
// Entire per-node pipeline lives in LDS.
// ---------------------------------------------------------------------------

typedef __attribute__((ext_vector_type(16))) _Float16 v16h;
typedef __attribute__((ext_vector_type(8)))  _Float16 v8h;
typedef __attribute__((ext_vector_type(8)))  float    v8f;

#define HIST   50
#define LP     64      // padded neighbor rows (50 -> 64 = 4 row tiles)
#define DIM    64
#define K2D    128

// A-fragment (16x32 f16): lane<16: row=lane,   K = [0..7]  and [16..23]
//                         lane>=16: row=lane-16, K = [8..15] and [24..31]
__device__ __forceinline__ v16h load_a_frag(const _Float16* base, int stride, int lane) {
  const int row = lane & 15;
  const int kb  = (lane >> 4) << 3;
  const _Float16* p = base + row * stride + kb;
  v16h f;
  ((v8h*)&f)[0] = *(const v8h*)(p);
  ((v8h*)&f)[1] = *(const v8h*)(p + 16);
  return f;
}

// B-fragment (32x16 f16), B[k][n] = W[n][k] (row-major W, row-stride = K):
//   lane<16: col n=lane,    K = [0..7]  and [16..23]
//   lane>=16: col n=lane-16, K = [8..15] and [24..31]
__device__ __forceinline__ v16h load_b_frag(const _Float16* wbase, int stride, int lane) {
  const int n  = lane & 15;
  const int kb = (lane >> 4) << 3;
  const _Float16* p = wbase + n * stride + kb;
  v16h f;
  ((v8h*)&f)[0] = *(const v8h*)(p);
  ((v8h*)&f)[1] = *(const v8h*)(p + 16);
  return f;
}

// One wave computes rows [rowbase, rowbase+16) x 64 cols of  relu(A @ W^T + b),
// storing f16 into D with row stride DSTRIDE. A row-stride == W row-stride == K.
template<int K, int DSTRIDE>
__device__ __forceinline__ void gemm16_relu(const _Float16* __restrict__ A,   // already offset to rowbase
                                            const _Float16* __restrict__ W,
                                            const float*    __restrict__ bias,
                                            _Float16*       __restrict__ D,
                                            int rowbase, int lane) {
  const int r0  = rowbase + ((lane >> 4) << 3);
  const int nlo = lane & 15;
#pragma unroll
  for (int nt = 0; nt < 4; ++nt) {
    v8f acc = {0.f, 0.f, 0.f, 0.f, 0.f, 0.f, 0.f, 0.f};
#pragma unroll
    for (int kt = 0; kt < K / 32; ++kt) {
      v16h av = load_a_frag(A + kt * 32, K, lane);
      v16h bv = load_b_frag(W + (nt * 16) * K + kt * 32, K, lane);
      acc = __builtin_amdgcn_wmma_f32_16x16x32_f16(false, av, false, bv,
                                                   (short)0, acc, false, false);
    }
    const int col = nt * 16 + nlo;
    const float bz = bias[col];
#pragma unroll
    for (int i = 0; i < 8; ++i) {
      float v = acc[i] + bz;            // D layout: VGPR i -> M = i (+8 for hi lanes)
      v = v > 0.f ? v : 0.f;
      D[(r0 + i) * DSTRIDE + col] = (_Float16)v;
    }
  }
}

__global__ __launch_bounds__(128)
void UserEncode_44839458570801_kernel(const int* __restrict__ nodes,
                                      const int* __restrict__ up_hist,
                                      const int* __restrict__ ur_hist,
                                      const float* __restrict__ u2e,
                                      const float* __restrict__ i2e,
                                      const float* __restrict__ r2e,
                                      const float* __restrict__ w1, const float* __restrict__ b1,
                                      const float* __restrict__ w2, const float* __restrict__ b2,
                                      const float* __restrict__ a1, const float* __restrict__ ab1,
                                      const float* __restrict__ a2, const float* __restrict__ ab2,
                                      const float* __restrict__ a3, const float* __restrict__ ab3,
                                      float* __restrict__ out) {
  // LDS (time-multiplexed, ~58 KB):
  __shared__ __align__(16) _Float16 s_io[LP * K2D];   // gather input -> later concat(o, u)
  __shared__ __align__(16) _Float16 s_x [LP * DIM];   // layer-1 out -> later att-layer-1 out
  __shared__ __align__(16) _Float16 s_h2[LP * DIM];   // att-layer-2 out
  __shared__ __align__(16) _Float16 s_wk[DIM * K2D];  // w1 -> later a1
  __shared__ __align__(16) _Float16 s_ws[DIM * DIM];  // w2 -> later a2
  __shared__ float s_b1[DIM], s_b2[DIM], s_ab1[DIM], s_ab2[DIM], s_a3[DIM];
  __shared__ _Float16 s_uh[DIM];
  __shared__ float s_logit[LP];
  __shared__ float s_att[LP];

  const int tid  = threadIdx.x;
  const int lane = tid & 31;
  const int wv   = tid >> 5;
  const int rowbase = wv * 16;
  const int node = blockIdx.x;

  // ---- Stage 0: gathers + weights into LDS (f32 -> f16) -------------------
  for (int i = tid; i < LP * K2D; i += 128) {
    const int l = i >> 7, c = i & 127;
    float v = 0.f;
    if (l < HIST) {
      if (c < DIM) v = i2e[up_hist[node * HIST + l] * DIM + c];
      else         v = r2e[ur_hist[node * HIST + l] * DIM + (c - DIM)];
    }
    s_io[i] = (_Float16)v;
  }
  for (int i = tid; i < DIM * K2D; i += 128) s_wk[i] = (_Float16)w1[i];
  for (int i = tid; i < DIM * DIM; i += 128) s_ws[i] = (_Float16)w2[i];
  if (tid < DIM) {
    s_b1[tid]  = b1[tid];
    s_b2[tid]  = b2[tid];
    s_ab1[tid] = ab1[tid];
    s_ab2[tid] = ab2[tid];
    s_a3[tid]  = a3[tid];
    s_uh[tid]  = (_Float16)u2e[nodes[node] * DIM + tid];
  }
  __syncthreads();

  // ---- GEMM1: x = relu(concat(p,r) @ w1^T + b1)   [64x128]x[128x64] -------
  gemm16_relu<K2D, DIM>(s_io + rowbase * K2D, s_wk, s_b1, s_x, rowbase, lane);
  __syncthreads();

  // ---- Refill: a1 -> s_wk, broadcast u into right half of s_io ------------
  for (int i = tid; i < DIM * K2D; i += 128) s_wk[i] = (_Float16)a1[i];
  for (int i = tid; i < LP * DIM; i += 128) {
    const int l = i >> 6, c = i & 63;
    s_io[l * K2D + DIM + c] = s_uh[c];
  }
  // ---- GEMM2: o = relu(x @ w2^T + b2) -> left half of s_io ----------------
  gemm16_relu<DIM, K2D>(s_x + rowbase * DIM, s_ws, s_b2, s_io, rowbase, lane);
  __syncthreads();

  // ---- Refill: a2 -> s_ws; GEMM3: relu(concat(o,u) @ a1^T + ab1) -> s_x ---
  for (int i = tid; i < DIM * DIM; i += 128) s_ws[i] = (_Float16)a2[i];
  gemm16_relu<K2D, DIM>(s_io + rowbase * K2D, s_wk, s_ab1, s_x, rowbase, lane);
  __syncthreads();

  // ---- GEMM4: relu(s_x @ a2^T + ab2) -> s_h2 ------------------------------
  gemm16_relu<DIM, DIM>(s_x + rowbase * DIM, s_ws, s_ab2, s_h2, rowbase, lane);
  __syncthreads();

  // ---- Logits: logit[l] = ab3 + s_h2[l,:] . a3 ----------------------------
  if (tid < LP) {
    float acc = ab3[0];
    for (int c = 0; c < DIM; ++c) acc += (float)s_h2[tid * DIM + c] * s_a3[c];
    s_logit[tid] = acc;
  }
  __syncthreads();

  // ---- Softmax over l < HIST (padded rows excluded) -----------------------
  if (tid < LP) {
    float m = -1e30f;
    for (int l = 0; l < HIST; ++l) m = fmaxf(m, s_logit[l]);
    float s = 0.f;
    for (int l = 0; l < HIST; ++l) s = s + __expf(s_logit[l] - m);
    s_att[tid] = (tid < HIST) ? (__expf(s_logit[tid] - m) / s) : 0.f;
  }
  __syncthreads();

  // ---- Weighted sum: out[d] = sum_l att[l] * o[l,d] -----------------------
  if (tid < DIM) {
    float acc = 0.f;
    for (int l = 0; l < HIST; ++l)
      acc += s_att[l] * (float)s_io[l * K2D + tid];
    out[node * DIM + tid] = acc;
  }
}

extern "C" void kernel_launch(void* const* d_in, const int* in_sizes, int n_in,
                              void* d_out, int out_size, void* d_ws, size_t ws_size,
                              hipStream_t stream) {
  const int*   nodes = (const int*)d_in[0];
  const int*   up    = (const int*)d_in[1];
  const int*   ur    = (const int*)d_in[2];
  // d_in[3] = pr_content (unused by the reference computation)
  const float* u2e = (const float*)d_in[4];
  const float* i2e = (const float*)d_in[5];
  const float* r2e = (const float*)d_in[6];
  const float* w1  = (const float*)d_in[7];
  const float* b1  = (const float*)d_in[8];
  const float* w2  = (const float*)d_in[9];
  const float* b2  = (const float*)d_in[10];
  const float* a1  = (const float*)d_in[11];
  const float* ab1 = (const float*)d_in[12];
  const float* a2  = (const float*)d_in[13];
  const float* ab2 = (const float*)d_in[14];
  const float* a3  = (const float*)d_in[15];
  const float* ab3 = (const float*)d_in[16];

  const int n_nodes = in_sizes[0];
  UserEncode_44839458570801_kernel<<<n_nodes, 128, 0, stream>>>(
      nodes, up, ur, u2e, i2e, r2e,
      w1, b1, w2, b2, a1, ab1, a2, ab2, a3, ab3,
      (float*)d_out);
}
